// X_Linear_Attention_4148938408122
// MI455X (gfx1250) — matrix operand order repaired and checked by reading, not compiled, not run on hardware
//
#include <hip/hip_runtime.h>
#include <math.h>

typedef __attribute__((ext_vector_type(16))) _Float16 v16h;
typedef __attribute__((ext_vector_type(8)))  _Float16 h8;
typedef __attribute__((ext_vector_type(8)))  float    v8f;

#define Sq   128
#define Dm   512
#define Hh   8
#define KDc  64
#define FBc  512
#define QT   16

// ---------------------------------------------------------------------------
// Kernel 1: Y = elu(X @ W + b), stored f16 in head-split layout [b][h][s][kd]
// X: (B*S, 512) fp32, W: (512,512), b: (512)
// grid = (B, S/16), block = 256
// ---------------------------------------------------------------------------
__global__ __launch_bounds__(256) void proj_elu_kernel(
    const float* __restrict__ X, const float* __restrict__ W,
    const float* __restrict__ bias, _Float16* __restrict__ out)
{
    __shared__ float sX[16][Dm];
    const int b = blockIdx.x, st = blockIdx.y, t = threadIdx.x;

    for (int i = t; i < 16 * Dm; i += 256) {
        int r = i >> 9, c = i & 511;
        sX[r][c] = X[((b * Sq) + st * 16 + r) * Dm + c];
    }
    __syncthreads();

    for (int cc = 0; cc < 2; ++cc) {
        const int f = t * 2 + cc;
        float acc[16];
#pragma unroll
        for (int r = 0; r < 16; ++r) acc[r] = 0.f;
        for (int d = 0; d < Dm; ++d) {
            float w = W[d * Dm + f];
#pragma unroll
            for (int r = 0; r < 16; ++r) acc[r] += sX[r][d] * w;
        }
        const float bv = bias[f];
        const int h = f >> 6, kd = f & 63;
#pragma unroll
        for (int r = 0; r < 16; ++r) {
            float y = acc[r] + bv;
            y = (y > 0.f) ? y : (__expf(y) - 1.f);     // elu
            int s = st * 16 + r;
            out[(((b * Hh + h) * Sq) + s) * KDc + kd] = (_Float16)y;
        }
    }
}

// ---------------------------------------------------------------------------
// Kernel 2: fused BK/logits/pool/softmax/channel/Bv, WMMA f16 core.
// grid = (B*H, S/QT), block = 256 (8 waves)
// ---------------------------------------------------------------------------
__global__ __launch_bounds__(256) void fused_attn_kernel(
    const _Float16* __restrict__ qk16, const _Float16* __restrict__ qv16,
    const _Float16* __restrict__ kk16, const _Float16* __restrict__ vv16,
    const float* __restrict__ amask,  const float* __restrict__ WKB,
    const float* __restrict__ bKB,    const float* __restrict__ Wb,
    const float* __restrict__ bb,     const float* __restrict__ We,
    const float* __restrict__ be,     float* __restrict__ O)
{
    __shared__ _Float16 sC [Sq * KDc];   // elu(kk)  row-major [k][d]
    __shared__ _Float16 sCV[Sq * KDc];   // elu(vv)
    __shared__ _Float16 sA [QT * KDc];   // elu(qk) for this q-tile
    __shared__ _Float16 sAV[QT * KDc];   // elu(qv)
    __shared__ float sBKB[FBc], sWbv[FBc];
    __shared__ float sPool[FBc];
    __shared__ float sLogits[Sq], sMask[Sq], sAttn[Sq];
    __shared__ float sRed[256], sChan[KDc], sScal[4];

    const int t     = threadIdx.x;
    const int w     = t >> 5;           // wave id 0..7
    const int lane  = t & 31;
    const int l16   = lane & 15;
    const int lhalf = lane >> 4;
    const int bh = blockIdx.x;
    const int b  = bh >> 3, h = bh & 7;
    const int q0 = blockIdx.y * QT;
    const float bbv = bb[0];

    // ---- stage per-(b,h) operands into LDS -------------------------------
    {
        const long base  = (long)(b * Hh + h) * Sq * KDc;
        for (int i = t; i < Sq * KDc; i += 256) {
            sC [i] = kk16[base + i];
            sCV[i] = vv16[base + i];
        }
        const long abase = base + (long)q0 * KDc;
        for (int i = t; i < QT * KDc; i += 256) {
            sA [i] = qk16[abase + i];
            sAV[i] = qv16[abase + i];
        }
        for (int i = t; i < FBc; i += 256) {
            sBKB[i] = bKB[i];
            sWbv[i] = Wb[i];
        }
    }

    // ---- load invariant B fragments of WKB (K=d x N=f) into registers ----
    // layout: lane holds column N=l16, K chunks kb*32 + {8*lhalf, 16+8*lhalf} .. +7
    v16h Bfrag[4][2];
#pragma unroll
    for (int j = 0; j < 4; ++j) {
        const int f = (w * 4 + j) * 16 + l16;
#pragma unroll
        for (int kb = 0; kb < 2; ++kb) {
#pragma unroll
            for (int e = 0; e < 16; ++e) {
                int d = kb * 32 + 8 * lhalf + (e & 7) + ((e >> 3) << 4);
                Bfrag[j][kb][e] = (_Float16)WKB[d * FBc + f];
            }
        }
    }
    __syncthreads();

    // ---- per-q loop -------------------------------------------------------
    for (int q = 0; q < QT; ++q) {
        // zero accumullators in LDS, load mask row
        if (t < Sq) {
            sLogits[t] = 0.f;
            sMask[t]   = amask[((long)b * Sq + (q0 + q)) * Sq + t];
        }
        sPool[t] = 0.f; sPool[t + 256] = 0.f;
        __syncthreads();
        if (t == 0) {                      // mask row sum (pool divisor)
            float ms = 0.f;
            for (int k = 0; k < Sq; ++k) ms += sMask[k];
            sScal[0] = ms;
        }

        // -------- WMMA core: C_q = (c .* a[q]) @ WKB, fused epilogue -------
        float poolacc[4] = {0.f, 0.f, 0.f, 0.f};
#pragma unroll 1
        for (int mt = 0; mt < 8; ++mt) {
            v8f acc[4];
#pragma unroll
            for (int j = 0; j < 4; ++j) acc[j] = (v8f){0,0,0,0,0,0,0,0};

#pragma unroll
            for (int kb = 0; kb < 2; ++kb) {
                // A fragment: row M=mt*16+l16, K chunks {8*lhalf, 16+8*lhalf}
                const int row = mt * 16 + l16;
                const int d0  = kb * 32 + 8 * lhalf;
                h8 c0 = *(const h8*)(sC + row * KDc + d0);
                h8 c1 = *(const h8*)(sC + row * KDc + d0 + 16);
                h8 a0 = *(const h8*)(sA + q   * KDc + d0);
                h8 a1 = *(const h8*)(sA + q   * KDc + d0 + 16);
                v16h cf = __builtin_shufflevector(c0, c1, 0,1,2,3,4,5,6,7,8,9,10,11,12,13,14,15);
                v16h af = __builtin_shufflevector(a0, a1, 0,1,2,3,4,5,6,7,8,9,10,11,12,13,14,15);
                v16h aa = cf * af;          // packed f16 scaling
#pragma unroll
                for (int j = 0; j < 4; ++j)
                    acc[j] = __builtin_amdgcn_wmma_f32_16x16x32_f16(
                        false, aa, false, Bfrag[j][kb], (short)0, acc[j], false, false);
            }

            // epilogue: relu(+bKB), pool += BK*mask, logits += BK*Wb
            float lgacc[8];
#pragma unroll
            for (int r = 0; r < 8; ++r) lgacc[r] = 0.f;
#pragma unroll
            for (int j = 0; j < 4; ++j) {
                const int f = (w * 4 + j) * 16 + l16;
                const float bkb = sBKB[f], wbf = sWbv[f];
#pragma unroll
                for (int r = 0; r < 8; ++r) {
                    float val = acc[j][r] + bkb;
                    val = fmaxf(val, 0.f);
                    int k = mt * 16 + lhalf * 8 + r;
                    poolacc[j] += val * sMask[k];
                    lgacc[r]   += val * wbf;
                }
            }
#pragma unroll
            for (int r = 0; r < 8; ++r) {   // sum the 16 f's of this half
                lgacc[r] += __shfl_xor(lgacc[r], 1);
                lgacc[r] += __shfl_xor(lgacc[r], 2);
                lgacc[r] += __shfl_xor(lgacc[r], 4);
                lgacc[r] += __shfl_xor(lgacc[r], 8);
            }
            if (l16 == 0) {
#pragma unroll
                for (int r = 0; r < 8; ++r)
                    atomicAdd(&sLogits[mt * 16 + lhalf * 8 + r], lgacc[r]);
            }
        }
#pragma unroll
        for (int j = 0; j < 4; ++j)
            atomicAdd(&sPool[(w * 4 + j) * 16 + l16], poolacc[j]);
        __syncthreads();

        // -------- softmax over k ------------------------------------------
        if (t < Sq) {
            float lv = sLogits[t] + bbv + (1.f - sMask[t]) * (-1e9f);
            sLogits[t] = lv;
            sRed[t] = lv;
        } else sRed[t] = -1e30f;
        __syncthreads();
        for (int stp = 128; stp >= 1; stp >>= 1) {
            if (t < stp) sRed[t] = fmaxf(sRed[t], sRed[t + stp]);
            __syncthreads();
        }
        const float mx = sRed[0];
        __syncthreads();
        if (t < Sq) { float e = __expf(sLogits[t] - mx); sAttn[t] = e; sRed[t] = e; }
        else sRed[t] = 0.f;
        __syncthreads();
        for (int stp = 128; stp >= 1; stp >>= 1) {
            if (t < stp) sRed[t] += sRed[t + stp];
            __syncthreads();
        }
        const float asum = sRed[0];
        const float msum = sScal[0];
        __syncthreads();

        // -------- channel = sigmoid(mean_k(BK) @ We + be) -----------------
        const int d = t & 63, part = t >> 6;
        {
            float partial = 0.f;
            const int f0 = part * 128;
            for (int f = f0; f < f0 + 128; ++f) partial += sPool[f] * We[f * KDc + d];
            sRed[t] = partial;
        }
        __syncthreads();
        if (t < KDc) {
            float tot = sRed[d] + sRed[d + 64] + sRed[d + 128] + sRed[d + 192];
            float x = tot / msum + be[d];
            sChan[d] = 1.f / (1.f + __expf(-x));
        }
        __syncthreads();

        // -------- out = ev[q,d] * sum_k evv[k,d]*attn[k] * chan[d] --------
        {
            float partial = 0.f;
            const int ks = part * 32;
            for (int k = ks; k < ks + 32; ++k)
                partial += (float)sCV[k * KDc + d] * sAttn[k];
            sRed[t] = partial;
        }
        __syncthreads();
        if (t < KDc) {
            float tot = sRed[d] + sRed[d + 64] + sRed[d + 128] + sRed[d + 192];
            float outv = (float)sAV[q * KDc + d] * (tot / asum) * sChan[d];
            O[((long)b * Sq + (q0 + q)) * Dm + h * KDc + d] = outv;
        }
        __syncthreads();
    }
}

// ---------------------------------------------------------------------------
// Kernel 3: out = O @ Wd + bd   (256x512 @ 512x64)
// ---------------------------------------------------------------------------
__global__ __launch_bounds__(256) void final_proj_kernel(
    const float* __restrict__ O, const float* __restrict__ Wd,
    const float* __restrict__ bd, float* __restrict__ out)
{
    const int i = blockIdx.x * 256 + threadIdx.x;     // 16384 outputs
    const int row = i >> 6, col = i & 63;
    float acc = 0.f;
    for (int dd = 0; dd < Dm; ++dd) acc += O[row * Dm + dd] * Wd[dd * KDc + col];
    out[i] = acc + bd[col];
}

// ---------------------------------------------------------------------------
extern "C" void kernel_launch(void* const* d_in, const int* in_sizes, int n_in,
                              void* d_out, int out_size, void* d_ws, size_t ws_size,
                              hipStream_t stream)
{
    const float* q    = (const float*)d_in[0];
    const float* v    = (const float*)d_in[1];
    const float* am   = (const float*)d_in[2];
    const float* Wkq  = (const float*)d_in[3];
    const float* bkq  = (const float*)d_in[4];
    const float* Wvq  = (const float*)d_in[5];
    const float* bvq  = (const float*)d_in[6];
    const float* Wk   = (const float*)d_in[7];
    const float* bk   = (const float*)d_in[8];
    const float* Wv   = (const float*)d_in[9];
    const float* bv   = (const float*)d_in[10];
    const float* WKB  = (const float*)d_in[11];
    const float* bKB  = (const float*)d_in[12];
    const float* Wb   = (const float*)d_in[13];
    const float* bb   = (const float*)d_in[14];
    const float* We   = (const float*)d_in[15];
    const float* be   = (const float*)d_in[16];
    const float* Wd   = (const float*)d_in[17];
    const float* bd   = (const float*)d_in[18];
    float* out = (float*)d_out;

    char* ws = (char*)d_ws;
    const size_t NHALF = (size_t)2 * Hh * Sq * KDc;       // 131072 halves
    _Float16* qk16 = (_Float16*)(ws);
    _Float16* qv16 = (_Float16*)(ws + 1 * NHALF * 2);
    _Float16* kk16 = (_Float16*)(ws + 2 * NHALF * 2);
    _Float16* vv16 = (_Float16*)(ws + 3 * NHALF * 2);
    float*    Obuf = (float*)   (ws + 4 * NHALF * 2);     // 2*128*512 fp32

    dim3 g1(2, Sq / 16);
    proj_elu_kernel<<<g1, 256, 0, stream>>>(q, Wkq, bkq, qk16);
    proj_elu_kernel<<<g1, 256, 0, stream>>>(q, Wvq, bvq, qv16);
    proj_elu_kernel<<<g1, 256, 0, stream>>>(v, Wk,  bk,  kk16);
    proj_elu_kernel<<<g1, 256, 0, stream>>>(v, Wv,  bv,  vv16);

    dim3 g2(2 * Hh, Sq / QT);
    fused_attn_kernel<<<g2, 256, 0, stream>>>(qk16, qv16, kk16, vv16,
                                              am, WKB, bKB, Wb, bb, We, be, Obuf);

    final_proj_kernel<<<(2 * Sq * KDc) / 256, 256, 0, stream>>>(Obuf, Wd, bd, out);
}
